// EncodeProcessDecode_85959475462362
// MI455X (gfx1250) — matrix operand order repaired and checked by reading, not compiled, not run on hardware
//
#include <hip/hip_runtime.h>
#include <cstdint>
#include <cstddef>

#define N_NODES 50000
#define N_EDGES 1600000
#define HIDDEN  128
#define NODE_IN 16
#define EDGE_IN 16

typedef __attribute__((ext_vector_type(16))) _Float16 v16h;
typedef __attribute__((ext_vector_type(8)))  float    v8f;
typedef __attribute__((ext_vector_type(4)))  unsigned uint4v;
typedef __attribute__((ext_vector_type(4)))  float    float4v;

// Fragment container: 16 f16 per lane = 8 VGPRs (A or B operand of 16x16x32 f16 WMMA)
union Frag16 {
    v16h     v;
    _Float16 h[16];
    uint4v   q[2];
};

// Order-preserving float<->uint encode so unsigned atomicMin == float min.
__device__ __forceinline__ unsigned enc_f32(float f) {
    unsigned u = __float_as_uint(f);
    return u ^ ((unsigned)((int)u >> 31) | 0x80000000u);
}
__device__ __forceinline__ float dec_f32(unsigned k) {
    unsigned u = (k & 0x80000000u) ? (k ^ 0x80000000u) : ~k;
    return __uint_as_float(u);
}

// ---------------------------------------------------------------------------
// Pack W [K_actual x 128] (f32, row-major) into WMMA B fragments (f16).
// Fragment (k_tile, n_tile): 32 lanes x 16 halves. Lane L holds column
// N = n*16 + (L%16), K = k*32 + (L/16)*8 + j (j<8), k*32 + 16 + (L/16)*8 + j-8 (j>=8).  K >= K_actual -> 0.
// grid.x = k_tiles, block = 256 (8 n_tiles x 32 lanes).
// ---------------------------------------------------------------------------
__global__ void pack_b_frags(const float* __restrict__ W, int K_actual,
                             _Float16* __restrict__ out) {
    const int k    = blockIdx.x;
    const int n    = threadIdx.x >> 5;
    const int lane = threadIdx.x & 31;
    _Float16* dst = out + (size_t)(((k * 8 + n) * 32 + lane) * 16);
    const int col   = n * 16 + (lane & 15);
    const int kbase = k * 32 + (lane >> 4) * 8;
#pragma unroll
    for (int j = 0; j < 16; ++j) {
        int K = kbase + (j < 8 ? j : j + 8);
        float v = (K < K_actual) ? W[(size_t)K * HIDDEN + col] : 0.0f;
        dst[j] = (_Float16)v;
    }
}

__global__ void init_aggr_kernel(unsigned* __restrict__ a, int n) {
    int i = blockIdx.x * blockDim.x + threadIdx.x;
    if (i < n) a[i] = 0xFFFFFFFFu;
}

// ---------------------------------------------------------------------------
// Encoder: h = relu(x @ W_enc + b_enc), output f16 [N_NODES][128].
// One wave per 16-node tile. A: K=0..15 from x, K=16..31 zero-padded.
// ---------------------------------------------------------------------------
__global__ void encoder_kernel(const float* __restrict__ x,
                               const _Float16* __restrict__ fragB,
                               const float* __restrict__ bias,
                               _Float16* __restrict__ h) {
    const int lane = threadIdx.x & 31;
    const int tile = blockIdx.x * (blockDim.x >> 5) + (threadIdx.x >> 5);
    if (tile >= N_NODES / 16) return;   // wave-uniform: EXEC stays all-ones
    const int m = lane & 15, half = lane >> 4;
    const int row0 = tile * 16;

    Frag16 a;
    const float* xr = x + (size_t)(row0 + m) * NODE_IN + half * 8;
    float4v x0 = ((const float4v*)xr)[0];
    float4v x1 = ((const float4v*)xr)[1];
    a.h[0] = (_Float16)x0.x; a.h[1] = (_Float16)x0.y;
    a.h[2] = (_Float16)x0.z; a.h[3] = (_Float16)x0.w;
    a.h[4] = (_Float16)x1.x; a.h[5] = (_Float16)x1.y;
    a.h[6] = (_Float16)x1.z; a.h[7] = (_Float16)x1.w;
#pragma unroll
    for (int j = 8; j < 16; ++j) a.h[j] = (_Float16)0.0f;

#pragma unroll
    for (int n = 0; n < 8; ++n) {
        Frag16 b;
        const uint4v* bq = (const uint4v*)(fragB + (size_t)((n * 32 + lane) * 16));
        b.q[0] = bq[0]; b.q[1] = bq[1];
        v8f c = {};
        c = __builtin_amdgcn_wmma_f32_16x16x32_f16(false, a.v, false, b.v,
                                                   (short)0, c, false, false);
        const float bb = bias[n * 16 + m];
#pragma unroll
        for (int r = 0; r < 8; ++r) {
            float v = c[r] + bb;
            v = v > 0.0f ? v : 0.0f;
            h[(size_t)(row0 + r + 8 * half) * HIDDEN + n * 16 + m] = (_Float16)v;
        }
    }
}

// ---------------------------------------------------------------------------
// Message + scatter-min (hot kernel): one wave per 16-edge tile.
// A[K=0..127]   : gathered h[src] rows (L2-resident), b128 loads map exactly
//                 onto the A-fragment 16-byte K-chunks.
// A[K=128..143] : edge_attr; A[K=144..159] zero pad. 5 K-tiles x 8 N-tiles WMMA.
// Result min-scattered into encoded-u32 aggr via global_atomic_min_u32.
// ---------------------------------------------------------------------------
__global__ __launch_bounds__(256) void message_kernel(
        const _Float16* __restrict__ h, const int* __restrict__ edge_index,
        const float* __restrict__ edge_attr, const _Float16* __restrict__ fragB,
        const float* __restrict__ bias, unsigned* __restrict__ aggr) {
    const int lane = threadIdx.x & 31;
    const int tile = blockIdx.x * 8 + (threadIdx.x >> 5);   // grid sized exactly
    const int m = lane & 15, half = lane >> 4;
    const int e0 = tile * 16;

    const int src = edge_index[e0 + m];
    const int* dst_idx = edge_index + N_EDGES;

    Frag16 a[5];
    const uint4v* hrow = (const uint4v*)(h + (size_t)src * HIDDEN);
#pragma unroll
    for (int kt = 0; kt < 4; ++kt) {
        a[kt].q[0] = hrow[kt * 4 + half];       // K chunk {0..7}+lane-half offset
        a[kt].q[1] = hrow[kt * 4 + 2 + half];   // K chunk {16..23}+offset
    }
    const float* ea = edge_attr + (size_t)(e0 + m) * EDGE_IN + half * 8;
    float4v e0v = ((const float4v*)ea)[0];
    float4v e1v = ((const float4v*)ea)[1];
    a[4].h[0] = (_Float16)e0v.x; a[4].h[1] = (_Float16)e0v.y;
    a[4].h[2] = (_Float16)e0v.z; a[4].h[3] = (_Float16)e0v.w;
    a[4].h[4] = (_Float16)e1v.x; a[4].h[5] = (_Float16)e1v.y;
    a[4].h[6] = (_Float16)e1v.z; a[4].h[7] = (_Float16)e1v.w;
#pragma unroll
    for (int j = 8; j < 16; ++j) a[4].h[j] = (_Float16)0.0f;

    unsigned rowp[8];
#pragma unroll
    for (int r = 0; r < 8; ++r)
        rowp[r] = (unsigned)dst_idx[e0 + r + 8 * half] * (unsigned)HIDDEN;

#pragma unroll
    for (int n = 0; n < 8; ++n) {
        v8f c = {};
#pragma unroll
        for (int kt = 0; kt < 5; ++kt) {
            Frag16 b;
            const uint4v* bq =
                (const uint4v*)(fragB + (size_t)(((kt * 8 + n) * 32 + lane) * 16));
            b.q[0] = bq[0]; b.q[1] = bq[1];
            c = __builtin_amdgcn_wmma_f32_16x16x32_f16(false, a[kt].v, false, b.v,
                                                       (short)0, c, false, false);
        }
        const float bb = bias[n * 16 + m];
        const int col = n * 16 + m;
#pragma unroll
        for (int r = 0; r < 8; ++r) {
            atomicMin(&aggr[rowp[r] + col], enc_f32(c[r] + bb));
        }
    }
}

// Decode encoded mins; untouched (0xFFFFFFFF, i.e. no incoming edge) -> 0.
__global__ void finalize_kernel(const unsigned* __restrict__ enc,
                                _Float16* __restrict__ out) {
    int i = blockIdx.x * blockDim.x + threadIdx.x;
    if (i >= N_NODES * HIDDEN) return;
    unsigned k = enc[i];
    float f = (k == 0xFFFFFFFFu) ? 0.0f : dec_f32(k);
    out[i] = (_Float16)f;
}

// ---------------------------------------------------------------------------
// Update + decoder fused: h2 = [h|aggr] @ W_upd + b_upd (8x8 WMMAs per tile),
// then out = sigmoid(h2 @ W_dec + b_dec) with an in-register column dot and a
// 16-lane shuffle reduction (rows live in lane groups of 16).
// ---------------------------------------------------------------------------
__global__ void update_decoder_kernel(const _Float16* __restrict__ h,
                                      const _Float16* __restrict__ ag,
                                      const _Float16* __restrict__ fragB,
                                      const float* __restrict__ bias,
                                      const float* __restrict__ Wdec,
                                      const float* __restrict__ bdec,
                                      float* __restrict__ out) {
    const int lane = threadIdx.x & 31;
    const int tile = blockIdx.x * (blockDim.x >> 5) + (threadIdx.x >> 5);
    if (tile >= N_NODES / 16) return;   // wave-uniform
    const int m = lane & 15, half = lane >> 4;
    const int row0 = tile * 16;

    Frag16 a[8];
    const uint4v* hrow = (const uint4v*)(h + (size_t)(row0 + m) * HIDDEN);
    const uint4v* arow = (const uint4v*)(ag + (size_t)(row0 + m) * HIDDEN);
#pragma unroll
    for (int kt = 0; kt < 4; ++kt) {
        a[kt].q[0]     = hrow[kt * 4 + half];
        a[kt].q[1]     = hrow[kt * 4 + 2 + half];
        a[kt + 4].q[0] = arow[kt * 4 + half];
        a[kt + 4].q[1] = arow[kt * 4 + 2 + half];
    }

    float partial[8] = {};
#pragma unroll
    for (int n = 0; n < 8; ++n) {
        v8f c = {};
#pragma unroll
        for (int kt = 0; kt < 8; ++kt) {
            Frag16 b;
            const uint4v* bq =
                (const uint4v*)(fragB + (size_t)(((kt * 8 + n) * 32 + lane) * 16));
            b.q[0] = bq[0]; b.q[1] = bq[1];
            c = __builtin_amdgcn_wmma_f32_16x16x32_f16(false, a[kt].v, false, b.v,
                                                       (short)0, c, false, false);
        }
        const float bb = bias[n * 16 + m];
        const float wd = Wdec[n * 16 + m];
#pragma unroll
        for (int r = 0; r < 8; ++r) partial[r] += (c[r] + bb) * wd;
    }

    const float bd = bdec[0];
#pragma unroll
    for (int r = 0; r < 8; ++r) {
        float s = partial[r];
        s += __shfl_xor(s, 1, 32);
        s += __shfl_xor(s, 2, 32);
        s += __shfl_xor(s, 4, 32);
        s += __shfl_xor(s, 8, 32);
        if (m == 0) {
            float z = s + bd;
            out[row0 + r + 8 * half] = 1.0f / (1.0f + __expf(-z));
        }
    }
}

// ---------------------------------------------------------------------------
extern "C" void kernel_launch(void* const* d_in, const int* in_sizes, int n_in,
                              void* d_out, int out_size, void* d_ws, size_t ws_size,
                              hipStream_t stream) {
    const float* x     = (const float*)d_in[0];
    const int*   ei    = (const int*)d_in[1];     // int32 (JAX x64 disabled)
    const float* eattr = (const float*)d_in[2];
    const float* W_enc = (const float*)d_in[3];
    const float* b_enc = (const float*)d_in[4];
    const float* W_msg = (const float*)d_in[5];
    const float* b_msg = (const float*)d_in[6];
    const float* W_upd = (const float*)d_in[7];
    const float* b_upd = (const float*)d_in[8];
    const float* W_dec = (const float*)d_in[9];
    const float* b_dec = (const float*)d_in[10];
    float* out = (float*)d_out;
    (void)in_sizes; (void)n_in; (void)out_size; (void)ws_size;

    uintptr_t base = (uintptr_t)d_ws;
    uintptr_t off = 0;
    auto take = [&](size_t bytes) -> uintptr_t {
        uintptr_t p = (base + off + 255) & ~(uintptr_t)255;
        off = (p - base) + bytes;
        return p;
    };
    _Float16* frag_enc = (_Float16*)take((size_t)1 * 8 * 32 * 16 * sizeof(_Float16));
    _Float16* frag_msg = (_Float16*)take((size_t)5 * 8 * 32 * 16 * sizeof(_Float16));
    _Float16* frag_upd = (_Float16*)take((size_t)8 * 8 * 32 * 16 * sizeof(_Float16));
    _Float16* h_f16    = (_Float16*)take((size_t)N_NODES * HIDDEN * sizeof(_Float16));
    unsigned* aggr_enc = (unsigned*)take((size_t)N_NODES * HIDDEN * sizeof(unsigned));
    _Float16* aggr_f16 = (_Float16*)take((size_t)N_NODES * HIDDEN * sizeof(_Float16));

    pack_b_frags<<<1, 256, 0, stream>>>(W_enc, NODE_IN, frag_enc);
    pack_b_frags<<<5, 256, 0, stream>>>(W_msg, HIDDEN + EDGE_IN, frag_msg);
    pack_b_frags<<<8, 256, 0, stream>>>(W_upd, 2 * HIDDEN, frag_upd);

    const int nagg = N_NODES * HIDDEN;
    init_aggr_kernel<<<(nagg + 255) / 256, 256, 0, stream>>>(aggr_enc, nagg);

    encoder_kernel<<<(N_NODES / 16 + 7) / 8, 256, 0, stream>>>(x, frag_enc, b_enc, h_f16);
    message_kernel<<<N_EDGES / 16 / 8, 256, 0, stream>>>(h_f16, ei, eattr, frag_msg,
                                                         b_msg, aggr_enc);
    finalize_kernel<<<(nagg + 255) / 256, 256, 0, stream>>>(aggr_enc, aggr_f16);
    update_decoder_kernel<<<(N_NODES / 16 + 7) / 8, 256, 0, stream>>>(
        h_f16, aggr_f16, frag_upd, b_upd, W_dec, b_dec, out);
}
